// GAT_66623532696010
// MI455X (gfx1250) — compile-verified, matrix operand order checked by reading, not run on hardware
//
#include <hip/hip_runtime.h>
#include <hip/hip_bf16.h>

// ---------------------------------------------------------------------------
// GAT layer for MI455X (gfx1250, wave32, WMMA).
// N nodes, F=128 in-features, 128 units, 8 heads (head dim 16).
// ---------------------------------------------------------------------------

#define FDIM   128
#define HEADS  8
#define HDIM   16    // FDIM / HEADS
#define WPAD   136   // padded LDS stride (bf16 elems): 17*16B rows, bank-spread
#define XPAD   136
#define STRIPS_PER_BLOCK 4

typedef __bf16 bf16_t;
typedef bf16_t v16bf __attribute__((ext_vector_type(16)));
typedef bf16_t v8bf  __attribute__((ext_vector_type(8)));
typedef float  v8f   __attribute__((ext_vector_type(8)));

__device__ __forceinline__ unsigned short f2bf_bits(float f) {
    unsigned u = __builtin_bit_cast(unsigned, f);
    u += 0x7fffu + ((u >> 16) & 1u);          // round-to-nearest-even
    return (unsigned short)(u >> 16);
}

// Build one 16-elem bf16 WMMA operand from two contiguous 16B LDS runs
// (K = kb..kb+7 in elems 0..7, K = kb+16..kb+23 in elems 8..15).
__device__ __forceinline__ v16bf load_op(const unsigned short* p) {
    v8bf lo = *(const v8bf*)(p);
    v8bf hi = *(const v8bf*)(p + 16);
    return __builtin_shufflevector(lo, hi, 0, 1, 2, 3, 4, 5, 6, 7,
                                           8, 9, 10, 11, 12, 13, 14, 15);
}

// ---------------------------------------------------------------------------
// Kernel 0: zero the accumulator surfaces.
// ---------------------------------------------------------------------------
__global__ void gat_init(float* __restrict__ out,
                         float* __restrict__ maxb,
                         float* __restrict__ sumb,
                         int N) {
    int i = blockIdx.x * blockDim.x + threadIdx.x;
    if (i < N * FDIM) out[i] = 0.0f;
    if (i < N * HEADS) { maxb[i] = 0.0f; sumb[i] = 0.0f; }  // scores >= 0 (ReLU'd Q,K)
}

// ---------------------------------------------------------------------------
// Kernel 1: fused Q/K/V node projections via WMMA bf16 (fp32 accumulate).
// Block = 256 threads = 8 waves; wave w owns output columns [16w, 16w+16).
// All three 128x128 weight matrices staged TRANSPOSED ([n][k], stride WPAD)
// in LDS once per block (~102 KB of the 320 KB/WGP), then the block sweeps
// STRIPS_PER_BLOCK 16-row strips. Per K-step each wave loads the A operand
// once and feeds 3 WMMAs (Q,K,V) -> 12 v_wmma per strip per wave, all
// operands via 16B ds_load_b128.
// ---------------------------------------------------------------------------
__global__ __launch_bounds__(256)
void gat_qkv_gemm(const float* __restrict__ x,
                  const float* __restrict__ Wq, const float* __restrict__ bq,
                  const float* __restrict__ Wk, const float* __restrict__ bk,
                  const float* __restrict__ Wv,
                  float* __restrict__ Qn, float* __restrict__ Kn,
                  float* __restrict__ Vn, int N) {
    __shared__ unsigned short sWt[3][FDIM * WPAD];  // bf16 bits, [n][k] transposed
    __shared__ unsigned short sX[16 * XPAD];        // bf16 bits, [m][k]

    const int tid = threadIdx.x;

    // Stage all three weight matrices, transposed, bf16. Global reads are
    // coalesced over n (r = k*128 + n is the flat index).
    const float* Wm[3] = {Wq, Wk, Wv};
    for (int i = tid; i < 3 * FDIM * FDIM; i += 256) {
        int mat = i >> 14;
        int r   = i & (FDIM * FDIM - 1);
        int k   = r >> 7;
        int n   = r & (FDIM - 1);
        sWt[mat][n * WPAD + k] = f2bf_bits(Wm[mat][r]);
    }

    const int lane  = tid & 31;            // wave32
    const int wave  = tid >> 5;
    const int n0    = wave * 16;           // this wave's output-column tile
    const int nl    = lane & 15;
    const int kb    = (lane >> 4) * 8;     // lanes 16-31 carry K+8 (ISA layout)
    const int mbase = (lane >> 4) * 8;     // C/D: lanes 16-31 -> M = 8+i
    const int nglob = n0 + nl;

    const float bqv = bq[nglob];
    const float bkv = bk[nglob];

    for (int s = 0; s < STRIPS_PER_BLOCK; ++s) {
        const int strip = blockIdx.x * STRIPS_PER_BLOCK + s;
        const int row0  = strip * 16;
        if (row0 >= N) break;              // uniform across block

        __syncthreads();  // s==0: weights staged; s>0: prior strip's reads done

        // Stage this strip's 16x128 x tile (zero-pad tail rows).
        for (int i = tid; i < 16 * FDIM; i += 256) {
            int m = i >> 7, k = i & (FDIM - 1);
            int gr = row0 + m;
            sX[m * XPAD + k] = f2bf_bits(gr < N ? x[(size_t)gr * FDIM + k] : 0.0f);
        }

        // Prefetch next strip's x rows (global_prefetch_b8), 128B per lane.
        if (tid < 64 && s + 1 < STRIPS_PER_BLOCK) {
            int pr = row0 + 16 + (tid >> 2);
            if (pr < N)
                __builtin_prefetch(&x[(size_t)pr * FDIM + (tid & 3) * 32], 0, 1);
        }

        __syncthreads();

        v8f c0 = {0.f, 0.f, 0.f, 0.f, 0.f, 0.f, 0.f, 0.f};
        v8f c1 = c0, c2 = c0;

        #pragma unroll
        for (int kk = 0; kk < FDIM; kk += 32) {
            // A: 16x32 bf16, lane = row M; shared by all three projections.
            v16bf a  = load_op(&sX[nl * XPAD + kk + kb]);
            v16bf b0 = load_op(&sWt[0][nglob * WPAD + kk + kb]);
            v16bf b1 = load_op(&sWt[1][nglob * WPAD + kk + kb]);
            v16bf b2 = load_op(&sWt[2][nglob * WPAD + kk + kb]);
            c0 = __builtin_amdgcn_wmma_f32_16x16x32_bf16(false, a, false, b0,
                                                         (short)0, c0, false, false);
            c1 = __builtin_amdgcn_wmma_f32_16x16x32_bf16(false, a, false, b1,
                                                         (short)0, c1, false, false);
            c2 = __builtin_amdgcn_wmma_f32_16x16x32_bf16(false, a, false, b2,
                                                         (short)0, c2, false, false);
        }

        // C/D layout: VGPR i -> M = mbase+i, N = nglob.
        #pragma unroll
        for (int i = 0; i < 8; ++i) {
            int m = row0 + mbase + i;
            if (m < N) {
                float q = c0[i] + bqv;  q = q > 0.0f ? q : 0.0f;   // ReLU(xWq+bq)
                float k = c1[i] + bkv;  k = k > 0.0f ? k : 0.0f;   // ReLU(xWk+bk)
                Qn[(size_t)m * FDIM + nglob] = q;
                Kn[(size_t)m * FDIM + nglob] = k;
                Vn[(size_t)m * FDIM + nglob] = c2[i];              // xWv (no bias)
            }
        }
    }
}

// ---------------------------------------------------------------------------
// Kernel 2: per-edge attention scores + segment max (int atomicMax on float
// bits — valid because scores are non-negative). One thread per edge.
// Edges [0,E) come from edge_index; [E, E+N) are the implicit self loops.
// ---------------------------------------------------------------------------
__global__ void gat_edge_scores(const float* __restrict__ Q,
                                const float* __restrict__ K,
                                const int* __restrict__ row,
                                const int* __restrict__ col,
                                int E, int N,
                                float* __restrict__ att,
                                float* __restrict__ maxb) {
    int e = blockIdx.x * blockDim.x + threadIdx.x;
    int Etot = E + N;
    if (e >= Etot) return;
    int r = (e < E) ? row[e] : (e - E);
    int c = (e < E) ? col[e] : (e - E);

    const float4* q4 = (const float4*)(Q + (size_t)r * FDIM);
    const float4* k4 = (const float4*)(K + (size_t)c * FDIM);

    #pragma unroll
    for (int h = 0; h < HEADS; ++h) {
        float acc = 0.0f;
        #pragma unroll
        for (int v = 0; v < HDIM / 4; ++v) {
            float4 qv = q4[h * (HDIM / 4) + v];
            float4 kv = k4[h * (HDIM / 4) + v];
            acc += qv.x * kv.x + qv.y * kv.y + qv.z * kv.z + qv.w * kv.w;
        }
        att[(size_t)e * HEADS + h] = acc;
        atomicMax((unsigned*)&maxb[r * HEADS + h], __float_as_uint(acc));
    }
}

// ---------------------------------------------------------------------------
// Kernel 3: exp, segment-sum, and UNNORMALIZED scatter of exp*V in one pass
// (normalization deferred to the node-level finalize — saves an edge pass).
// One thread per (edge, head): 1 sum atomic + 16 value atomics.
// ---------------------------------------------------------------------------
__global__ void gat_edge_scatter(const float* __restrict__ V,
                                 const int* __restrict__ row,
                                 const int* __restrict__ col,
                                 int E, int N,
                                 const float* __restrict__ att,
                                 const float* __restrict__ maxb,
                                 float* __restrict__ sumb,
                                 float* __restrict__ out) {
    int idx = blockIdx.x * blockDim.x + threadIdx.x;
    int Etot = E + N;
    if (idx >= Etot * HEADS) return;
    int e = idx >> 3;
    int h = idx & (HEADS - 1);
    int r = (e < E) ? row[e] : (e - E);
    int c = (e < E) ? col[e] : (e - E);

    float m    = maxb[r * HEADS + h];
    float sexp = __expf(att[(size_t)e * HEADS + h] - m);
    atomicAdd(&sumb[r * HEADS + h], sexp);

    const float4* v4 = (const float4*)(V + (size_t)c * FDIM + h * HDIM);
    float* o = out + (size_t)r * FDIM + h * HDIM;
    #pragma unroll
    for (int v = 0; v < HDIM / 4; ++v) {
        float4 vv = v4[v];
        atomicAdd(&o[v * 4 + 0], sexp * vv.x);
        atomicAdd(&o[v * 4 + 1], sexp * vv.y);
        atomicAdd(&o[v * 4 + 2], sexp * vv.z);
        atomicAdd(&o[v * 4 + 3], sexp * vv.w);
    }
}

// ---------------------------------------------------------------------------
// Kernel 4: node-level finalize: divide by segment sum, add output bias.
// ---------------------------------------------------------------------------
__global__ void gat_finalize(float* __restrict__ out,
                             const float* __restrict__ sumb,
                             const float* __restrict__ bias,
                             int N) {
    int i = blockIdx.x * blockDim.x + threadIdx.x;
    if (i >= N * FDIM) return;
    int n = i >> 7;
    int j = i & (FDIM - 1);
    int h = j >> 4;
    out[i] = out[i] / sumb[n * HEADS + h] + bias[j];
}

// ---------------------------------------------------------------------------
// Host-side launcher.
// ---------------------------------------------------------------------------
extern "C" void kernel_launch(void* const* d_in, const int* in_sizes, int n_in,
                              void* d_out, int out_size, void* d_ws, size_t ws_size,
                              hipStream_t stream) {
    const float* x    = (const float*)d_in[0];
    const int*   ei   = (const int*)d_in[1];     // [2, E]: row then col
    const float* Wq   = (const float*)d_in[2];
    const float* bq   = (const float*)d_in[3];
    const float* Wk   = (const float*)d_in[4];
    const float* bk   = (const float*)d_in[5];
    const float* Wv   = (const float*)d_in[6];
    const float* bout = (const float*)d_in[7];
    float* out = (float*)d_out;

    const int N    = in_sizes[0] / FDIM;
    const int E    = in_sizes[1] / 2;
    const int Etot = E + N;
    const int* row = ei;
    const int* col = ei + E;

    // Workspace layout (floats): Qn | Kn | Vn | att | maxb | sumb  (~107 MB)
    float* Qn   = (float*)d_ws;
    float* Kn   = Qn + (size_t)N * FDIM;
    float* Vn   = Kn + (size_t)N * FDIM;
    float* att  = Vn + (size_t)N * FDIM;
    float* maxb = att + (size_t)Etot * HEADS;
    float* sumb = maxb + (size_t)N * HEADS;

    const int TB = 256;

    // 0) zero accumulators
    gat_init<<<(N * FDIM + TB - 1) / TB, TB, 0, stream>>>(out, maxb, sumb, N);

    // 1) Q/K/V projections (WMMA bf16); 4 row-strips per block
    int nstrips = (N + 15) / 16;
    int gblocks = (nstrips + STRIPS_PER_BLOCK - 1) / STRIPS_PER_BLOCK;
    gat_qkv_gemm<<<gblocks, 256, 0, stream>>>(x, Wq, bq, Wk, bk, Wv, Qn, Kn, Vn, N);

    // 2) attention scores + segment max
    gat_edge_scores<<<(Etot + TB - 1) / TB, TB, 0, stream>>>(Qn, Kn, row, col, E, N,
                                                             att, maxb);

    // 3) exp + segment sum + unnormalized V scatter
    int nscatter = Etot * HEADS;
    gat_edge_scatter<<<(nscatter + TB - 1) / TB, TB, 0, stream>>>(Vn, row, col, E, N,
                                                                  att, maxb, sumb, out);

    // 4) normalize + bias
    gat_finalize<<<(N * FDIM + TB - 1) / TB, TB, 0, stream>>>(out, sumb, bout, N);
}
